// ChemGeomFeatEncoder_52604759442023
// MI455X (gfx1250) — compile-verified, matrix operand order checked by reading, not run on hardware
//
#include <hip/hip_runtime.h>
#include <hip/hip_bf16.h>
#include <math.h>

// ---------------------------------------------------------------------------
// Problem constants (from reference)
// ---------------------------------------------------------------------------
#define H    128
#define GF   128
#define SF   64
#define CFS  144          // GF + 16 edge-feature width
#define CFSP 160          // padded to multiple of 32 for bf16 WMMA K
#define NN   20000
#define NV   50000
#define NE   800000
#define EPSV 1e-5f

typedef __bf16 bf16_t;
typedef __attribute__((ext_vector_type(16))) __bf16 v16bf;
typedef __attribute__((ext_vector_type(8)))  __bf16 v8bf;
typedef __attribute__((ext_vector_type(8)))  float  v8f;

union V16U { v16bf v; v8bf h[2]; };

__device__ inline v16bf mk16(v8bf lo, v8bf hi) { V16U u; u.h[0] = lo; u.h[1] = hi; return u.v; }

// Load one 16x32 bf16 WMMA fragment from a row-major [16 x ldk] bf16 tile.
// Per CDNA5 16-bit A layout: lane 0-15 -> row=lane, K chunks {k0+0..7, k0+16..23};
// lane 16-31 -> row=lane-16, K chunks {k0+8..15, k0+24..31}. B fragments use the
// same pattern with "row" = output column (weights stored K-major / transposed).
__device__ inline v16bf load_frag(const bf16_t* base, int ldk, int k0, int lane) {
  int r  = lane & 15;
  int kb = k0 + ((lane >> 4) << 3);
  v8bf lo = *(const v8bf*)(base + (size_t)r * ldk + kb);
  v8bf hi = *(const v8bf*)(base + (size_t)r * ldk + kb + 16);
  return mk16(lo, hi);
}

__device__ inline v8f wmma_bf16(v16bf a, v16bf b, v8f c) {
  // D = A(16x32 bf16) x B(32x16 bf16) + C(16x16 f32)
  return __builtin_amdgcn_wmma_f32_16x16x32_bf16(false, a, false, b, (short)0, c, false, false);
}

__device__ inline float siluf(float x)     { return x / (1.f + __expf(-x)); }
__device__ inline float sigmf(float x)     { return 1.f / (1.f + __expf(-x)); }
__device__ inline float softplusf(float x) { return (x > 20.f) ? x : log1pf(__expf(x)); }

// Generic->LDS: on AMDGPU the low 32 bits of a generic pointer into LDS are the
// LDS byte offset (aperture base lives in the high half).
__device__ inline unsigned lds_addr_of(const void* p) {
  return (unsigned)(unsigned long long)p;
}

// CDNA5 async copy: 8 bytes per lane, global -> LDS, tracked by ASYNCcnt.
// No VGPR destination; fence with s_wait_asynccnt before consuming the LDS data.
__device__ inline void async_copy_b64(unsigned lds_byte_addr, const void* gaddr) {
  asm volatile("global_load_async_to_lds_b64 %0, %1, off"
               :: "v"(lds_byte_addr), "v"(gaddr) : "memory");
}
__device__ inline void wait_asynccnt0() {
  asm volatile("s_wait_asynccnt 0x0" ::: "memory");
}

// ---------------------------------------------------------------------------
// BatchNorm-folding weight prep:  W1t[n][k] = w1[k][n] * s1[n], C1[n] = (b1-m1)*s1 + be1
// Weights stored K-major (transposed) so B fragments are contiguous 128-bit loads.
// ---------------------------------------------------------------------------
__global__ void fold_kernel(
    const float* w1, const float* b1, const float* g1, const float* be1, const float* m1, const float* v1,
    const float* w2, const float* b2, const float* g2, const float* be2, const float* m2, const float* v2,
    int K1, int K1p, int N1, int N2,
    bf16_t* W1t, float* C1, bf16_t* W2t, float* C2)
{
  size_t nC1 = (size_t)N1, nC2 = (size_t)N2;
  size_t nW1 = (size_t)K1p * N1, nW2 = (size_t)N1 * N2;
  size_t total = nC1 + nC2 + nW1 + nW2;
  size_t stride = (size_t)gridDim.x * blockDim.x;
  for (size_t i = (size_t)blockIdx.x * blockDim.x + threadIdx.x; i < total; i += stride) {
    if (i < nC1) {
      int n = (int)i;
      float s = g1[n] * rsqrtf(v1[n] + EPSV);
      C1[n] = (b1[n] - m1[n]) * s + be1[n];
    } else if (i < nC1 + nC2) {
      int n = (int)(i - nC1);
      float s = g2[n] * rsqrtf(v2[n] + EPSV);
      C2[n] = (b2[n] - m2[n]) * s + be2[n];
    } else if (i < nC1 + nC2 + nW1) {
      size_t q = i - nC1 - nC2;
      int n = (int)(q / K1p), k = (int)(q % K1p);
      float s = g1[n] * rsqrtf(v1[n] + EPSV);
      W1t[(size_t)n * K1p + k] = (k < K1) ? (bf16_t)(w1[(size_t)k * N1 + n] * s) : (bf16_t)0.f;
    } else {
      size_t q = i - nC1 - nC2 - nW1;
      int n = (int)(q / N1), k = (int)(q % N1);
      float s = g2[n] * rsqrtf(v2[n] + EPSV);
      W2t[(size_t)n * N1 + k] = (bf16_t)(w2[(size_t)k * N2 + n] * s);
    }
  }
}

__global__ void cast_f32_bf16_kernel(const float* __restrict__ in, bf16_t* __restrict__ out, size_t n) {
  size_t stride = (size_t)gridDim.x * blockDim.x;
  for (size_t i = (size_t)blockIdx.x * blockDim.x + threadIdx.x; i < n; i += stride)
    out[i] = (bf16_t)in[i];
}

__global__ void zero_f32_kernel(float* __restrict__ p, size_t n) {
  size_t stride = (size_t)gridDim.x * blockDim.x;
  for (size_t i = (size_t)blockIdx.x * blockDim.x + threadIdx.x; i < n; i += stride)
    p[i] = 0.f;
}

// Cast rows x cols f32 into a strided bf16 destination (left half of concat buffer).
__global__ void cast_rows_kernel(const float* __restrict__ in, bf16_t* __restrict__ out,
                                 int rows, int cols, int ldo) {
  size_t n = (size_t)rows * cols;
  size_t stride = (size_t)gridDim.x * blockDim.x;
  for (size_t i = (size_t)blockIdx.x * blockDim.x + threadIdx.x; i < n; i += stride) {
    int r = (int)(i / cols), c = (int)(i % cols);
    out[(size_t)r * ldo + c] = (bf16_t)in[i];
  }
}

// ---------------------------------------------------------------------------
// Generic fused 2-layer MLP:  out = silu(X @ W1' + c1) @ W2' + c2
// One wave per 16 rows; 8 waves / block. N1 = N2 = 128.
// ---------------------------------------------------------------------------
template <int K1, bool OUT_BF16>
__global__ __launch_bounds__(256) void mlp2_kernel(
    const bf16_t* __restrict__ X, int ldx,
    const bf16_t* __restrict__ W1t, const float* __restrict__ C1,
    const bf16_t* __restrict__ W2t, const float* __restrict__ C2,
    int M, float* __restrict__ outF, bf16_t* __restrict__ outB, int ldo)
{
  __shared__ __align__(16) bf16_t hid[8][16 * H];
  int lane = threadIdx.x & 31;
  int wv   = threadIdx.x >> 5;
  int row0 = (blockIdx.x * 8 + wv) * 16;
  if (row0 >= M) return;

  // Warm the WGP$ for layer-2 weights while layer 1 runs.
  __builtin_prefetch(W2t + (size_t)lane * 256, 0, 3);

  int mOff = (lane >> 4) << 3;   // C-layout: lanes 16-31 hold rows M+8..M+15
  int cl   = lane & 15;
  bf16_t* hD = hid[wv];

  // Resident A fragments for layer 1 (rows are 16-divisible; M % 16 == 0 here).
  v16bf a1[K1 / 32];
#pragma unroll
  for (int ks = 0; ks < K1 / 32; ++ks)
    a1[ks] = load_frag(X + (size_t)row0 * ldx, ldx, ks * 32, lane);

#pragma unroll
  for (int t = 0; t < 8; ++t) {          // 8 column tiles of 16 -> N1 = 128
    v8f acc = {};
#pragma unroll
    for (int ks = 0; ks < K1 / 32; ++ks) {
      v16bf b = load_frag(W1t + (size_t)(t * 16) * K1, K1, ks * 32, lane);
      acc = wmma_bf16(a1[ks], b, acc);
    }
    float bias = C1[t * 16 + cl];
#pragma unroll
    for (int r = 0; r < 8; ++r)
      hD[(r + mOff) * H + t * 16 + cl] = (bf16_t)siluf(acc[r] + bias);
  }

  // Layer 2 (K = 128)
  v16bf a2[H / 32];
#pragma unroll
  for (int ks = 0; ks < H / 32; ++ks)
    a2[ks] = load_frag(hD, H, ks * 32, lane);

#pragma unroll
  for (int t = 0; t < 8; ++t) {
    v8f acc = {};
#pragma unroll
    for (int ks = 0; ks < H / 32; ++ks) {
      v16bf b = load_frag(W2t + (size_t)(t * 16) * H, H, ks * 32, lane);
      acc = wmma_bf16(a2[ks], b, acc);
    }
    float bias = C2[t * 16 + cl];
#pragma unroll
    for (int r = 0; r < 8; ++r) {
      float v  = acc[r] + bias;
      int row  = row0 + r + mOff;
      if (OUT_BF16) outB[(size_t)row * ldo + t * 16 + cl] = (bf16_t)v;
      else          outF[(size_t)row * ldo + t * 16 + cl] = v;
    }
  }
}

// ---------------------------------------------------------------------------
// Edge pipeline: build edge_feats (144 -> padded 160), 2-layer MLP (160->128->256),
// msg = sigmoid(s[:, :128]) * softplus(s[:, 128:]), atomic segment-sum into agg.
// One wave = 16 edges; 4 waves / block. The chem-row gather uses the CDNA5
// async global->LDS path (no VGPR round-trip, ASYNCcnt-tracked).
// ---------------------------------------------------------------------------
__global__ __launch_bounds__(128) void edge_kernel(
    const bf16_t* __restrict__ gxbf,
    const float* __restrict__ node_pos,
    const float* __restrict__ vnormals,
    const float* __restrict__ verts,
    const float* __restrict__ dist,
    const int* __restrict__ nbr_vids,
    const int* __restrict__ vert_nbr_ind,
    const bf16_t* __restrict__ W1t, const float* __restrict__ C1,   // [128][160], [128]
    const bf16_t* __restrict__ W2t, const float* __restrict__ C2,   // [256][128], [256]
    float* __restrict__ agg)
{
  __shared__ __align__(16) bf16_t atile[4][16 * CFSP];
  __shared__ __align__(16) bf16_t hid[4][16 * H];
  int lane = threadIdx.x & 31;
  int wv   = threadIdx.x >> 5;
  int row0 = (blockIdx.x * 4 + wv) * 16;
  if (row0 >= NE) return;

  bf16_t* A  = atile[wv];
  bf16_t* hD = hid[wv];

  // Warm the WGP$ for layer-2 weights (256x128 bf16) while we build features.
  __builtin_prefetch(W2t + (size_t)lane * 1024, 0, 3);

  // Gather chem rows via async global->LDS: one instruction per edge moves
  // 32 lanes x 8 B = the full 256 B bf16 row into the LDS A-tile.
  {
    unsigned ldsA = lds_addr_of(A) + (unsigned)lane * 8u;
    for (int e = 0; e < 16; ++e) {
      int nid = vert_nbr_ind[row0 + e];
      const char* ga = (const char*)(gxbf + (size_t)nid * GF) + (size_t)lane * 8u;
      async_copy_b64(ldsA + (unsigned)e * (CFSP * 2), ga);
    }
  }

  // Per-edge scalar features on lanes 0..15 (distinct LDS columns; no overlap
  // with the async-written chem columns).
  if (lane < 16) {
    int e   = lane;
    int eid = row0 + e;
    float d = dist[eid];
    const float dw = 8.f / 7.f;   // linspace(0,8,8) step == width
#pragma unroll
    for (int i = 0; i < 8; ++i) {
      float x = (d - dw * i) / dw;
      A[e * CFSP + GF + i] = (bf16_t)__expf(-x * x);
    }
    int nid = vert_nbr_ind[eid];
    int vid = nbr_vids[eid];
    float vx = node_pos[nid * 3 + 0] - verts[vid * 3 + 0];
    float vy = node_pos[nid * 3 + 1] - verts[vid * 3 + 1];
    float vz = node_pos[nid * 3 + 2] - verts[vid * 3 + 2];
    float inv = rsqrtf(vx * vx + vy * vy + vz * vz);
    float ang = inv * (vx * vnormals[vid * 3 + 0] + vy * vnormals[vid * 3 + 1] + vz * vnormals[vid * 3 + 2]);
    const float aw = 2.f / 7.f;   // linspace(-1,1,8)
#pragma unroll
    for (int i = 0; i < 8; ++i) {
      float x = (ang - (-1.f + aw * i)) / aw;
      A[e * CFSP + GF + 8 + i] = (bf16_t)__expf(-x * x);
    }
  }
  // Zero the 16 pad columns (144..159): 32 lanes x 8 bf16.
  {
    int e = lane >> 1;
    ((uint4*)(A + e * CFSP + CFS))[lane & 1] = make_uint4(0, 0, 0, 0);
  }

  // Fence the async gather before reading A-tile fragments.
  wait_asynccnt0();

  int mOff = (lane >> 4) << 3;
  int cl   = lane & 15;

  // Layer 1: K = 160 (5 k-steps), N = 128 (8 col tiles), silu, hidden -> LDS bf16.
  v16bf a1[5];
#pragma unroll
  for (int ks = 0; ks < 5; ++ks)
    a1[ks] = load_frag(A, CFSP, ks * 32, lane);

#pragma unroll
  for (int t = 0; t < 8; ++t) {
    v8f acc = {};
#pragma unroll
    for (int ks = 0; ks < 5; ++ks) {
      v16bf b = load_frag(W1t + (size_t)(t * 16) * CFSP, CFSP, ks * 32, lane);
      acc = wmma_bf16(a1[ks], b, acc);
    }
    float bias = C1[t * 16 + cl];
#pragma unroll
    for (int r = 0; r < 8; ++r)
      hD[(r + mOff) * H + t * 16 + cl] = (bf16_t)siluf(acc[r] + bias);
  }

  // Layer 2: compute filter col-tile t and core col-tile 128+t pairwise,
  // msg = sigmoid(f) * softplus(c), scatter-add into agg (L2-resident, f32 atomics).
  v16bf a2[4];
#pragma unroll
  for (int ks = 0; ks < 4; ++ks)
    a2[ks] = load_frag(hD, H, ks * 32, lane);

  int vid_[8];
#pragma unroll
  for (int r = 0; r < 8; ++r)
    vid_[r] = nbr_vids[row0 + r + mOff];

#pragma unroll
  for (int t = 0; t < 8; ++t) {
    v8f af = {}, ac = {};
#pragma unroll
    for (int ks = 0; ks < 4; ++ks) {
      v16bf bf_ = load_frag(W2t + (size_t)(t * 16) * H, H, ks * 32, lane);
      v16bf bc_ = load_frag(W2t + (size_t)((128 + t * 16)) * H, H, ks * 32, lane);
      af = wmma_bf16(a2[ks], bf_, af);
      ac = wmma_bf16(a2[ks], bc_, ac);
    }
    float bf1 = C2[t * 16 + cl];
    float bc1 = C2[128 + t * 16 + cl];
#pragma unroll
    for (int r = 0; r < 8; ++r) {
      float m = sigmf(af[r] + bf1) * softplusf(ac[r] + bc1);
      atomicAdd(&agg[(size_t)vid_[r] * H + t * 16 + cl], m);
    }
  }
}

// ---------------------------------------------------------------------------
// Host launcher
// ---------------------------------------------------------------------------
extern "C" void kernel_launch(void* const* d_in, const int* in_sizes, int n_in,
                              void* d_out, int out_size, void* d_ws, size_t ws_size,
                              hipStream_t stream)
{
  (void)in_sizes; (void)n_in; (void)out_size; (void)ws_size;

  const float* graph_x      = (const float*)d_in[0];
  const float* node_pos     = (const float*)d_in[1];
  const float* surf_x       = (const float*)d_in[2];
  const float* vnormals     = (const float*)d_in[3];
  const float* verts        = (const float*)d_in[4];
  const float* vdist        = (const float*)d_in[5];
  const int*   nbr_vids     = (const int*)d_in[6];
  const int*   vert_nbr_ind = (const int*)d_in[7];
  const float* const* P     = (const float* const*)d_in;
  // param dicts flatten in insertion order: w1,b1,g1,be1,m1,v1,w2,b2,g2,be2,m2,v2
  // chem_p: 8..19, surf_chem_p: 20..31, geom_p: 32..43, feat_p: 44..55

  char* ws = (char*)d_ws;
  size_t off = 0;
  auto alloc = [&](size_t bytes) -> char* {
    char* p = ws + off;
    off += (bytes + 255) & ~(size_t)255;
    return p;
  };

  bf16_t* gxbf = (bf16_t*)alloc((size_t)NN * GF * 2);      // graph_x in bf16 (gather table + A)
  bf16_t* sxbf = (bf16_t*)alloc((size_t)NV * SF * 2);      // surf_x in bf16
  float*  agg  = (float*) alloc((size_t)NV * H * 4);       // segment_sum accumulator
  bf16_t* cat  = (bf16_t*)alloc((size_t)NV * 2 * H * 2);   // [agg_bf | h_geom_bf] concat

  bf16_t* cW1 = (bf16_t*)alloc(128 * 128 * 2); float* cC1 = (float*)alloc(128 * 4);
  bf16_t* cW2 = (bf16_t*)alloc(128 * 128 * 2); float* cC2 = (float*)alloc(128 * 4);
  bf16_t* sW1 = (bf16_t*)alloc(128 * 160 * 2); float* sC1 = (float*)alloc(128 * 4);
  bf16_t* sW2 = (bf16_t*)alloc(256 * 128 * 2); float* sC2 = (float*)alloc(256 * 4);
  bf16_t* gW1 = (bf16_t*)alloc(128 *  64 * 2); float* gC1 = (float*)alloc(128 * 4);
  bf16_t* gW2 = (bf16_t*)alloc(128 * 128 * 2); float* gC2 = (float*)alloc(128 * 4);
  bf16_t* fW1 = (bf16_t*)alloc(128 * 256 * 2); float* fC1 = (float*)alloc(128 * 4);
  bf16_t* fW2 = (bf16_t*)alloc(128 * 128 * 2); float* fC2 = (float*)alloc(128 * 4);

  float* out_hgeom = (float*)d_out;                  // (NV, 128)
  float* out_chem  = out_hgeom + (size_t)NV * H;     // (NN, 128)

  // Fold BN into weights; transpose to K-major bf16.
  fold_kernel<<<64, 256, 0, stream>>>(P[8],  P[9],  P[10], P[11], P[12], P[13],
                                      P[14], P[15], P[16], P[17], P[18], P[19],
                                      GF, GF, H, H, cW1, cC1, cW2, cC2);
  fold_kernel<<<96, 256, 0, stream>>>(P[20], P[21], P[22], P[23], P[24], P[25],
                                      P[26], P[27], P[28], P[29], P[30], P[31],
                                      CFS, CFSP, H, 2 * H, sW1, sC1, sW2, sC2);
  fold_kernel<<<64, 256, 0, stream>>>(P[32], P[33], P[34], P[35], P[36], P[37],
                                      P[38], P[39], P[40], P[41], P[42], P[43],
                                      SF, SF, H, H, gW1, gC1, gW2, gC2);
  fold_kernel<<<96, 256, 0, stream>>>(P[44], P[45], P[46], P[47], P[48], P[49],
                                      P[50], P[51], P[52], P[53], P[54], P[55],
                                      2 * H, 2 * H, H, H, fW1, fC1, fW2, fC2);

  cast_f32_bf16_kernel<<<2048, 256, 0, stream>>>(graph_x, gxbf, (size_t)NN * GF);
  cast_f32_bf16_kernel<<<2048, 256, 0, stream>>>(surf_x,  sxbf, (size_t)NV * SF);
  zero_f32_kernel<<<2048, 256, 0, stream>>>(agg, (size_t)NV * H);

  // chem_graph = MLP(graph_x)  -> d_out second part (f32)
  mlp2_kernel<GF, false><<<(NN + 127) / 128, 256, 0, stream>>>(
      gxbf, GF, cW1, cC1, cW2, cC2, NN, out_chem, nullptr, H);

  // h_geom = MLP(surf_x) -> right half of concat buffer (bf16, stride 256)
  mlp2_kernel<SF, true><<<(NV + 127) / 128, 256, 0, stream>>>(
      sxbf, SF, gW1, gC1, gW2, gC2, NV, nullptr, cat + H, 2 * H);

  // Edge features + edge MLP + gated message + atomic segment_sum.
  edge_kernel<<<NE / 64, 128, 0, stream>>>(
      gxbf, node_pos, vnormals, verts, vdist, nbr_vids, vert_nbr_ind,
      sW1, sC1, sW2, sC2, agg);

  // agg -> left half of concat buffer (bf16).
  cast_rows_kernel<<<2048, 256, 0, stream>>>(agg, cat, NV, H, 2 * H);

  // h_geom = MLP(concat[agg, h_geom]) -> d_out first part (f32)
  mlp2_kernel<2 * H, false><<<(NV + 127) / 128, 256, 0, stream>>>(
      cat, 2 * H, fW1, fC1, fW2, fC2, NV, out_hgeom, nullptr, H);
}
